// MultiHeadAttention_47768626266719
// MI455X (gfx1250) — matrix operand order repaired and checked by reading, not compiled, not run on hardware
//
#include <hip/hip_runtime.h>
#include <hip/hip_bf16.h>

// ---------------------------------------------------------------------------
// MHA forward for MI455X (gfx1250, wave32, WMMA, async global->LDS DMA).
// Pipeline: [f32->f16 cvt of X and weights] -> [QKV gemm, head-split f16] ->
//           [flash attention, WMMA] -> [out-proj gemm -> f32].
// All GEMM/attention tiles are staged with GLOBAL_LOAD_ASYNC_TO_LDS_B128
// into double-buffered LDS so DMA overlaps WMMA compute.
// ---------------------------------------------------------------------------

typedef __attribute__((ext_vector_type(16))) _Float16 v16h;
typedef __attribute__((ext_vector_type(4)))  _Float16 v4h;
typedef __attribute__((ext_vector_type(8)))  float    v8f;

constexpr int Bc  = 4;     // batch
constexpr int Sc  = 2048;  // seq
constexpr int Hc  = 1024;  // hidden
constexpr int NHc = 16;    // heads
constexpr int HDc = 64;    // head dim

// ---------------- CDNA5 async global->LDS copy (ASYNCcnt tracked) ----------

__device__ __forceinline__ void async_copy_b128(uint32_t lds_off,
                                                const void* gaddr) {
  asm volatile("global_load_async_to_lds_b128 %0, %1, off"
               :
               : "v"(lds_off), "v"((unsigned long long)(uintptr_t)gaddr)
               : "memory");
}

__device__ __forceinline__ void wait_async0() {
  asm volatile("s_wait_asynccnt 0x0" ::: "memory");
}

// ---------------- WMMA helpers (contiguous k order: k = 16*(lane/16) + e in A and B) ----

__device__ __forceinline__ v8f wmma_f16(v16h a, v16h b, v8f c) {
  return __builtin_amdgcn_wmma_f32_16x16x32_f16(
      /*neg_a=*/false, a, /*neg_b=*/false, b,
      /*c_mod=*/(short)0, c, /*reuse_a=*/false, /*reuse_b=*/false);
}

// A fragment 16x32 (MxK) from LDS row-major tile, stride lda (f16 elems).
__device__ __forceinline__ v16h load_A_frag(const _Float16* lds, int lda,
                                            int row0, int k0) {
  const int lane = threadIdx.x & 31;
  const int row  = row0 + (lane & 15);
  const int kb   = (lane >> 4) ? 16 : 0;
  v16h a;
#pragma unroll
  for (int e = 0; e < 16; ++e) {
    const int k = kb + e;
    a[e] = lds[row * lda + k0 + k];
  }
  return a;
}

// B fragment 32x16 (KxN) where LDS holds B transposed: lds[n*ldb + k].
__device__ __forceinline__ v16h load_B_frag_nk(const _Float16* lds, int ldb,
                                               int n0, int k0) {
  const int lane = threadIdx.x & 31;
  const int n    = n0 + (lane & 15);
  const int kb   = (lane >> 4) ? 16 : 0;
  v16h b;
#pragma unroll
  for (int e = 0; e < 16; ++e) b[e] = lds[n * ldb + k0 + kb + e];
  return b;
}

// B fragment 32x16 (KxN) where LDS holds B row-major: lds[k*ldn + n].
__device__ __forceinline__ v16h load_B_frag_kn(const _Float16* lds, int ldn,
                                               int k0, int n0) {
  const int lane = threadIdx.x & 31;
  const int n    = n0 + (lane & 15);
  const int kb   = (lane >> 4) ? 16 : 0;
  v16h b;
#pragma unroll
  for (int e = 0; e < 16; ++e) b[e] = lds[(k0 + kb + e) * ldn + n];
  return b;
}

// ---------------- f32 -> f16 conversion (one-shot, bandwidth bound) --------

__global__ __launch_bounds__(256) void cvt_f32_f16_kernel(
    const float* __restrict__ in, _Float16* __restrict__ out, int n4) {
  const int i = blockIdx.x * 256 + threadIdx.x;
  if (i < n4) {
    const float4 v = ((const float4*)in)[i];
    v4h h;
    h[0] = (_Float16)v.x; h[1] = (_Float16)v.y;
    h[2] = (_Float16)v.z; h[3] = (_Float16)v.w;
    ((v4h*)out)[i] = h;
  }
}

// ---------------- GEMM: Out[m,n] = sum_k A[m,k] * W[n,k] + bias[n] ---------
// A, W already f16. Tile 128x128, BK=32, double-buffered async LDS staging.
// 256 threads = 8 waves in 4x2; each wave computes 32x64 (8 WMMA / K-step).
template <bool SPLIT_HEADS, bool OUT_F32>
__global__ __launch_bounds__(256) void gemm_xwT_kernel(
    const _Float16* __restrict__ A, const _Float16* __restrict__ W,
    const float* __restrict__ bias, void* __restrict__ Out,
    int M, int N, int K) {
  constexpr int BM = 128, BN = 128, BK = 32;
  __shared__ __align__(16) _Float16 sA[2][BM * BK];  // [m][k]
  __shared__ __align__(16) _Float16 sB[2][BN * BK];  // [n][k] (B col-major)

  const int t    = threadIdx.x;
  const int w    = t >> 5;
  const int lane = t & 31;
  const int hi   = lane >> 4;
  const int ln   = lane & 15;
  const int wm   = (w >> 1) * 32;   // wave row offset
  const int wn   = (w & 1) * 64;    // wave col offset
  const int m0   = blockIdx.y * BM;
  const int n0   = blockIdx.x * BN;

  // 8KB per tile = 512 x 16B chunks; each thread issues 2 A + 2 B copies.
  auto issue_tile = [&](int buf, int kk) {
#pragma unroll
    for (int i = 0; i < 2; ++i) {
      const int chunk = i * 256 + t;
      const int row = chunk >> 2, c = chunk & 3;
      async_copy_b128((uint32_t)(uintptr_t)&sA[buf][row * BK + c * 8],
                      A + (size_t)(m0 + row) * K + kk + c * 8);
      async_copy_b128((uint32_t)(uintptr_t)&sB[buf][row * BK + c * 8],
                      W + (size_t)(n0 + row) * K + kk + c * 8);
    }
  };

  v8f acc[2][4];
#pragma unroll
  for (int i = 0; i < 2; ++i)
#pragma unroll
    for (int j = 0; j < 4; ++j) acc[i][j] = {};

  issue_tile(0, 0);
  wait_async0();
  __syncthreads();

  for (int kk = 0; kk < K; kk += BK) {
    const int buf = (kk / BK) & 1;
    if (kk + BK < K) issue_tile(buf ^ 1, kk + BK);  // overlap DMA w/ compute

    const v16h a0 = load_A_frag(sA[buf], BK, wm, 0);
    const v16h a1 = load_A_frag(sA[buf], BK, wm + 16, 0);
#pragma unroll
    for (int j = 0; j < 4; ++j) {
      const v16h b = load_B_frag_nk(sB[buf], BK, wn + 16 * j, 0);
      acc[0][j] = wmma_f16(a0, b, acc[0][j]);
      acc[1][j] = wmma_f16(a1, b, acc[1][j]);
    }
    wait_async0();
    __syncthreads();
  }

  // --- epilogue: bias + store ---
#pragma unroll
  for (int mi = 0; mi < 2; ++mi) {
#pragma unroll
    for (int j = 0; j < 4; ++j) {
#pragma unroll
      for (int r = 0; r < 8; ++r) {
        const int gm = m0 + wm + mi * 16 + r + hi * 8;
        const int gn = n0 + wn + 16 * j + ln;
        const float val = acc[mi][j][r] + bias[gn];
        if (SPLIT_HEADS) {
          // [M=B*S, N=H] -> [B, NH, S, HD] f16
          const int bb = gm >> 11, ss = gm & (Sc - 1);
          const int hh = gn >> 6,  dd = gn & (HDc - 1);
          ((_Float16*)Out)[(((size_t)bb * NHc + hh) * Sc + ss) * HDc + dd] =
              (_Float16)val;
        } else if (OUT_F32) {
          ((float*)Out)[(size_t)gm * N + gn] = val;
        } else {
          ((_Float16*)Out)[(size_t)gm * N + gn] = (_Float16)val;
        }
      }
    }
  }
}

// ---------------- Flash attention: one block per (b, head, 64-query tile) --
// 128 threads = 4 waves; wave w owns query rows [16w, 16w+16).
// K/V tiles double-buffered, loaded by async DMA overlapped with compute.
__global__ __launch_bounds__(128) void attn_kernel(
    const _Float16* __restrict__ Qh, const _Float16* __restrict__ Kh,
    const _Float16* __restrict__ Vh, const float* __restrict__ mask,
    _Float16* __restrict__ Ctx) {
  __shared__ __align__(16) _Float16 sQ[64 * HDc];
  __shared__ __align__(16) _Float16 sK[2][64 * HDc];
  __shared__ __align__(16) _Float16 sV[2][64 * HDc];
  __shared__ __align__(16) _Float16 sP[4 * 16 * 64];

  const int t     = threadIdx.x;
  const int w     = t >> 5;
  const int lane  = t & 31;
  const int hi    = lane >> 4;
  const int ln    = lane & 15;
  const int qtile = blockIdx.x;
  const int head  = blockIdx.y;
  const int bz    = blockIdx.z;

  const size_t hbase = ((size_t)bz * NHc + head) * Sc * HDc;
  const _Float16* Qp = Qh + hbase;
  const _Float16* Kp = Kh + hbase;
  const _Float16* Vp = Vh + hbase;
  const float* maskRow = mask + (size_t)bz * Sc;

  // each K/V tile is a contiguous 8KB block = 512 x 16B chunks, 4 per thread
  auto issue_kv = [&](int buf, int kt) {
#pragma unroll
    for (int i = 0; i < 4; ++i) {
      const int chunk = i * 128 + t;
      async_copy_b128((uint32_t)(uintptr_t)&sK[buf][chunk * 8],
                      Kp + (size_t)kt * 64 * HDc + chunk * 8);
      async_copy_b128((uint32_t)(uintptr_t)&sV[buf][chunk * 8],
                      Vp + (size_t)kt * 64 * HDc + chunk * 8);
    }
  };

  // prologue: stage Q tile + first K/V tile
#pragma unroll
  for (int i = 0; i < 4; ++i) {
    const int chunk = i * 128 + t;
    async_copy_b128((uint32_t)(uintptr_t)&sQ[chunk * 8],
                    Qp + (size_t)qtile * 64 * HDc + chunk * 8);
  }
  issue_kv(0, 0);
  wait_async0();
  __syncthreads();

  const v16h qa0 = load_A_frag(sQ, HDc, 16 * w, 0);
  const v16h qa1 = load_A_frag(sQ, HDc, 16 * w, 32);

  v8f oacc[4];
#pragma unroll
  for (int j = 0; j < 4; ++j) oacc[j] = {};
  float mrow[8], lrow[8];
#pragma unroll
  for (int r = 0; r < 8; ++r) { mrow[r] = -1e30f; lrow[r] = 0.f; }

  _Float16* sPw = sP + w * 16 * 64;
  const float scale = 0.125f;  // 1/sqrt(64)
  constexpr int NT = Sc / 64;

  for (int kt = 0; kt < NT; ++kt) {
    const int p = kt & 1;
    if (kt + 1 < NT) issue_kv(p ^ 1, kt + 1);  // overlap DMA with compute

    // scores S = Q K^T * scale + mask  (16 rows x 64 keys per wave)
    v8f sfrag[4];
#pragma unroll
    for (int j = 0; j < 4; ++j) {
      v8f s = {};
      s = wmma_f16(qa0, load_B_frag_nk(sK[p], HDc, 16 * j, 0), s);
      s = wmma_f16(qa1, load_B_frag_nk(sK[p], HDc, 16 * j, 32), s);
      const float mv = maskRow[kt * 64 + 16 * j + ln];
#pragma unroll
      for (int r = 0; r < 8; ++r) s[r] = s[r] * scale + mv;
      sfrag[j] = s;
    }

    // row max: per-lane over j, then half-wave butterfly (rows live in a
    // 16-lane half with col = lane&15, matching the C-frag layout)
    float pm[8];
#pragma unroll
    for (int r = 0; r < 8; ++r) {
      float v = sfrag[0][r];
      v = fmaxf(v, sfrag[1][r]);
      v = fmaxf(v, sfrag[2][r]);
      v = fmaxf(v, sfrag[3][r]);
      v = fmaxf(v, __shfl_xor(v, 1, 16));
      v = fmaxf(v, __shfl_xor(v, 2, 16));
      v = fmaxf(v, __shfl_xor(v, 4, 16));
      v = fmaxf(v, __shfl_xor(v, 8, 16));
      pm[r] = v;
    }

    float fac[8], psum[8];
#pragma unroll
    for (int r = 0; r < 8; ++r) {
      const float nm = fmaxf(mrow[r], pm[r]);
      fac[r] = __expf(mrow[r] - nm);
      mrow[r] = nm;
      psum[r] = 0.f;
    }
#pragma unroll
    for (int j = 0; j < 4; ++j) {
#pragma unroll
      for (int r = 0; r < 8; ++r) {
        const float pv = __expf(sfrag[j][r] - mrow[r]);
        sfrag[j][r] = pv;
        psum[r] += pv;
      }
    }
#pragma unroll
    for (int r = 0; r < 8; ++r) {
      float v = psum[r];
      v += __shfl_xor(v, 1, 16);
      v += __shfl_xor(v, 2, 16);
      v += __shfl_xor(v, 4, 16);
      v += __shfl_xor(v, 8, 16);
      lrow[r] = lrow[r] * fac[r] + v;
#pragma unroll
      for (int j = 0; j < 4; ++j) oacc[j][r] *= fac[r];
    }

    // bounce P through wave-private LDS (in-order per wave -> no barrier)
#pragma unroll
    for (int j = 0; j < 4; ++j)
#pragma unroll
      for (int r = 0; r < 8; ++r)
        sPw[(r + hi * 8) * 64 + 16 * j + ln] = (_Float16)sfrag[j][r];

    const v16h pa0 = load_A_frag(sPw, 64, 0, 0);
    const v16h pa1 = load_A_frag(sPw, 64, 0, 32);
#pragma unroll
    for (int j = 0; j < 4; ++j) {
      oacc[j] = wmma_f16(pa0, load_B_frag_kn(sV[p], HDc, 0, 16 * j), oacc[j]);
      oacc[j] = wmma_f16(pa1, load_B_frag_kn(sV[p], HDc, 32, 16 * j), oacc[j]);
    }

    wait_async0();     // next tile's DMA landed
    __syncthreads();   // all waves done reading buffer p
  }

  // normalize and store ctx in [B, S, H] f16 layout (ready for out-proj)
#pragma unroll
  for (int r = 0; r < 8; ++r) {
    const float inv = 1.0f / lrow[r];
    const int q = qtile * 64 + w * 16 + r + hi * 8;
#pragma unroll
    for (int j = 0; j < 4; ++j) {
      Ctx[((size_t)bz * Sc + q) * Hc + head * HDc + 16 * j + ln] =
          (_Float16)(oacc[j][r] * inv);
    }
  }
}

// ---------------------------------------------------------------------------
extern "C" void kernel_launch(void* const* d_in, const int* in_sizes, int n_in,
                              void* d_out, int out_size, void* d_ws,
                              size_t ws_size, hipStream_t stream) {
  (void)in_sizes; (void)n_in; (void)out_size; (void)ws_size;
  const float* X    = (const float*)d_in[0];
  const float* mask = (const float*)d_in[1];
  const float* Wq   = (const float*)d_in[2];
  const float* bq   = (const float*)d_in[3];
  const float* Wk   = (const float*)d_in[4];
  const float* bk   = (const float*)d_in[5];
  const float* Wv   = (const float*)d_in[6];
  const float* bv   = (const float*)d_in[7];
  const float* Wo   = (const float*)d_in[8];
  const float* bo   = (const float*)d_in[9];

  const int M = Bc * Sc;                 // 8192
  const size_t perQKV = (size_t)M * Hc;  // 8,388,608 f16 elems
  const size_t perW   = (size_t)Hc * Hc; // 1,048,576 f16 elems

  _Float16* Xh  = (_Float16*)d_ws;       // f16 copies (workspace, ~92 MB)
  _Float16* Wqh = Xh  + perQKV;
  _Float16* Wkh = Wqh + perW;
  _Float16* Wvh = Wkh + perW;
  _Float16* Woh = Wvh + perW;
  _Float16* Qh  = Woh + perW;
  _Float16* Kh  = Qh  + perQKV;
  _Float16* Vh  = Kh  + perQKV;
  _Float16* Ctx = Vh  + perQKV;

  // one-shot f32 -> f16 conversions
  const int n4X = (int)(perQKV / 4), n4W = (int)(perW / 4);
  cvt_f32_f16_kernel<<<n4X / 256, 256, 0, stream>>>(X,  Xh,  n4X);
  cvt_f32_f16_kernel<<<n4W / 256, 256, 0, stream>>>(Wq, Wqh, n4W);
  cvt_f32_f16_kernel<<<n4W / 256, 256, 0, stream>>>(Wk, Wkh, n4W);
  cvt_f32_f16_kernel<<<n4W / 256, 256, 0, stream>>>(Wv, Wvh, n4W);
  cvt_f32_f16_kernel<<<n4W / 256, 256, 0, stream>>>(Wo, Woh, n4W);

  dim3 gGemm(Hc / 128, M / 128);
  // QKV projections: head-split f16 output
  gemm_xwT_kernel<true, false><<<gGemm, 256, 0, stream>>>(Xh, Wqh, bq, Qh, M, Hc, Hc);
  gemm_xwT_kernel<true, false><<<gGemm, 256, 0, stream>>>(Xh, Wkh, bk, Kh, M, Hc, Hc);
  gemm_xwT_kernel<true, false><<<gGemm, 256, 0, stream>>>(Xh, Wvh, bv, Vh, M, Hc, Hc);

  // attention
  dim3 gAttn(Sc / 64, NHc, Bc);
  attn_kernel<<<gAttn, 128, 0, stream>>>(Qh, Kh, Vh, mask, Ctx);

  // output projection: f16 ctx -> f32 out (row-major [B,S,H])
  gemm_xwT_kernel<false, true><<<gGemm, 256, 0, stream>>>(Ctx, Woh, bo, d_out, M, Hc, Hc);
}
